// GATLayer_11673721111209
// MI455X (gfx1250) — compile-verified, matrix operand order checked by reading, not run on hardware
//
#include <hip/hip_runtime.h>
#include <math.h>

// GAT layer, fused for MI455X (gfx1250, wave32, WMMA bf16 K=32).
// Sizes fixed by the reference harness.
constexpr int B_ = 8;
constexpr int N_ = 2048;
constexpr int D_ = 256;
#define ALPHA_LRELU 0.2f

typedef __attribute__((ext_vector_type(16))) __bf16 v16bf;
typedef __attribute__((ext_vector_type(8)))  float  v8f;

union ABPack {
    uint4          u4[2];
    unsigned int   u[8];
    unsigned short s[16];
    v16bf          v;
};

__device__ __forceinline__ unsigned short f2bf(float f) {
    unsigned int u = __float_as_uint(f);
    u += 0x7FFFu + ((u >> 16) & 1u);            // round-to-nearest-even
    return (unsigned short)(u >> 16);
}
__device__ __forceinline__ float bf2f(unsigned short s) {
    return __uint_as_float(((unsigned int)s) << 16);
}
#if __has_builtin(__builtin_amdgcn_cvt_pk_bf16_f32)
typedef __attribute__((ext_vector_type(2))) __bf16 v2bf;
__device__ __forceinline__ unsigned int pack2bf(float a, float b) {
    union { v2bf v; unsigned int u; } t;
    t.v = __builtin_amdgcn_cvt_pk_bf16_f32(a, b);
    return t.u;
}
#else
__device__ __forceinline__ unsigned int pack2bf(float a, float b) {
    return (unsigned int)f2bf(a) | ((unsigned int)f2bf(b) << 16);
}
#endif

// ---------------------------------------------------------------------------
// K0a: h fp32 -> hbf bf16, row-major, 8 elements/thread (one-shot, HBM-bound).
// Keeps all fp32->bf16 conversion OUT of the GEMM inner loops.
__global__ void convert_h_kernel(const float* __restrict__ h,
                                 unsigned short* __restrict__ hbf) {
    int idx = blockIdx.x * blockDim.x + threadIdx.x;   // B*N*D/8 threads
    const float4* p = (const float4*)(h + (size_t)idx * 8);
    float4 f0 = p[0], f1 = p[1];
    uint4 o;
    o.x = pack2bf(f0.x, f0.y);
    o.y = pack2bf(f0.z, f0.w);
    o.z = pack2bf(f1.x, f1.y);
    o.w = pack2bf(f1.z, f1.w);
    ((uint4*)hbf)[idx] = o;
}

// K0b: W [Din,D] fp32 -> WbfT [D,Din] bf16 (transposed so WMMA B-operand
// loads are contiguous b128s).
__global__ void convert_wt_kernel(const float* __restrict__ W,
                                  unsigned short* __restrict__ WbfT) {
    int idx = blockIdx.x * blockDim.x + threadIdx.x;   // 65536
    int n = idx >> 8, k = idx & 255;
    WbfT[idx] = f2bf(W[k * D_ + n]);                   // WbfT[n][k] = W[k][n]
}

// ---------------------------------------------------------------------------
// K1: Wh = h @ W via v_wmma_f32_16x16x32_bf16. One workgroup = 16 waves =
// one 16-row tile covering all 256 columns (one 16x16 tile per wave).
// A and B operands are pure b128 loads (pre-converted bf16). Output stored
// TRANSPOSED bf16: WhbfT[b][col][row] for the aggregation kernel's B operand.
__global__ __launch_bounds__(512) void wh_gemm_kernel(
        const unsigned short* __restrict__ hbf,
        const unsigned short* __restrict__ WbfT,
        unsigned short* __restrict__ WhbfT) {
    int bid  = blockIdx.x;                 // 0..1023
    int b    = bid >> 7;
    int i0   = (bid & 127) << 4;
    int lane = threadIdx.x & 31;
    int wave = threadIdx.x >> 5;           // 0..15
    int lm   = lane & 15;
    int hi   = lane >> 4;
    int kbase = hi ? 8 : 0;                // 16-bit A/B VGPR layout K-base
    int ncol  = wave * 16 + lm;

    v8f acc = {};
    const unsigned short* arow = hbf  + ((size_t)(b * N_ + i0 + lm)) * D_;
    const unsigned short* bcol = WbfT + (size_t)ncol * D_;

    for (int k0 = 0; k0 < D_; k0 += 32) {
        ABPack A, Bp;
        // dwords 0..3 = K kbase..kbase+7, dwords 4..7 = K kbase+16..kbase+23
        // (ISA 16-bit A/B 16x32 VGPR layout) -> two contiguous 16B loads each.
        const unsigned short* ap = arow + k0 + kbase;
        A.u4[0] = ((const uint4*)ap)[0];
        A.u4[1] = ((const uint4*)(ap + 16))[0];
        const unsigned short* bp = bcol + k0 + kbase;
        Bp.u4[0] = ((const uint4*)bp)[0];
        Bp.u4[1] = ((const uint4*)(bp + 16))[0];
        acc = __builtin_amdgcn_wmma_f32_16x16x32_bf16(
                  false, A.v, false, Bp.v, (short)0, acc, false, false);
    }
    // C/D layout: vgpr r -> row M = r + 8*hi, col N = lane%16. Consecutive r
    // are consecutive rows -> one 16B packed-bf16 store per lane (transposed).
    unsigned short* op = WhbfT + ((size_t)(b * D_ + ncol)) * N_ + i0 + hi * 8;
    uint4 ov;
    unsigned int* ow = (unsigned int*)&ov;
#pragma unroll
    for (int t = 0; t < 4; ++t)
        ow[t] = pack2bf(acc[2 * t], acc[2 * t + 1]);
    *((uint4*)op) = ov;
}

// ---------------------------------------------------------------------------
// K2: e_src/e_dst = Wh @ a[:D] / a[D:]. Reads transposed Wh -> coalesced
// across lanes (consecutive i), 'a' loads are wave-uniform (scalar path).
__global__ void edge_dot_kernel(const unsigned short* __restrict__ WhbfT,
                                const float* __restrict__ a,
                                float* __restrict__ e_src,
                                float* __restrict__ e_dst) {
    int idx = blockIdx.x * blockDim.x + threadIdx.x;   // B*N
    int b = idx >> 11, i = idx & (N_ - 1);
    const unsigned short* base = WhbfT + (size_t)b * D_ * N_ + i;
    float es = 0.f, ed = 0.f;
    for (int c = 0; c < D_; ++c) {
        float v = bf2f(base[(size_t)c * N_]);
        es += v * a[c];
        ed += v * a[D_ + c];
    }
    e_src[idx] = es;
    e_dst[idx] = ed;
}

// ---------------------------------------------------------------------------
// K3: per-row softmax stats (max m, sum l of exp(e-m)) over masked
// leakyrelu(e_src_i + e_dst_j). One wave32 per row; coalesced adj reads;
// butterfly __shfl_xor merge of (m,l) pairs.
__global__ void rowstat_kernel(const int* __restrict__ adj,
                               const float* __restrict__ e_src,
                               const float* __restrict__ e_dst,
                               float* __restrict__ mrow,
                               float* __restrict__ lrow) {
    int wid  = (blockIdx.x * blockDim.x + threadIdx.x) >> 5;  // row id 0..B*N-1
    int lane = threadIdx.x & 31;
    int b = wid >> 11, i = wid & (N_ - 1);
    float es = e_src[wid];
    const int*   arow = adj   + (size_t)i * N_;
    const float* ed   = e_dst + (size_t)b * N_;
    float m = -__builtin_inff(), l = 0.f;
    for (int j = lane; j < N_; j += 32) {
        if (arow[j] > 0) {
            float e = es + ed[j];
            e = e > 0.f ? e : ALPHA_LRELU * e;
            if (e > m) { l *= __expf(m - e); m = e; }
            l += __expf(e - m);
        }
    }
    for (int off = 16; off; off >>= 1) {
        float mo = __shfl_xor(m, off, 32);
        float lo = __shfl_xor(l, off, 32);
        float mn = fmaxf(m, mo);
        float s0 = (m == mn)  ? 1.f : __expf(m - mn);   // guards -inf - -inf
        float s1 = (mo == mn) ? 1.f : __expf(mo - mn);
        l = l * s0 + lo * s1;
        m = mn;
    }
    if (lane == 0) { mrow[wid] = m; lrow[wid] = l; }
}

// ---------------------------------------------------------------------------
// K4: fused masked-softmax @ Wh + ELU. One workgroup (16 waves) per 16-row
// tile; loop over j in macro-chunks of 64 (2x WMMA K). Each thread computes
// TWO P entries; one barrier publishes a 16x64 bf16 tile; each wave issues
// TWO back-to-back WMMAs from it (0.5 barriers per WMMA). Ping-pong across
// two tile sets keeps correctness with a single barrier per macro-chunk.
// Wh values streamed as WMMA B via contiguous global b128 (L2-resident),
// with global_prefetch of the next macro-chunk.
__global__ __launch_bounds__(512) void gat_aggregate_kernel(
        const unsigned short* __restrict__ WhbfT,
        const int* __restrict__ adj,
        const float* __restrict__ e_src,
        const float* __restrict__ e_dst,
        const float* __restrict__ mrow,
        const float* __restrict__ lrow,
        float* __restrict__ out) {
    __shared__ unsigned short pTile[2][16 * 80]; // 2 sets x (16 rows x 64 +pad)
    __shared__ float sm[16], sl[16], ses[16];

    int bid = blockIdx.x;
    int b   = bid >> 7;
    int i0  = (bid & 127) << 4;
    int tid = threadIdx.x;
    if (tid < 16) {
        sm[tid]  = mrow[b * N_ + i0 + tid];
        sl[tid]  = lrow[b * N_ + i0 + tid];
        ses[tid] = e_src[b * N_ + i0 + tid];
    }
    __syncthreads();

    int lane = tid & 31, wave = tid >> 5;
    int lm = lane & 15, hi = lane >> 4;
    int kbase = hi ? 8 : 0;
    int ncol  = wave * 16 + lm;
    int pr = tid >> 5;          // P-tile row this thread fills (0..15)
    int pj = tid & 31;          // P-tile col pair this thread fills

    float my_es = ses[pr];
    float my_m  = sm[pr];
    const float* edb  = e_dst + (size_t)b * N_;
    const int*   arow = adj + (size_t)(i0 + pr) * N_ + pj;
    const unsigned short* bcol = WhbfT + ((size_t)(b * D_ + ncol)) * N_;

    v8f acc = {};
    for (int j0 = 0; j0 < N_; j0 += 64) {
        int set = (j0 >> 6) & 1;
        // Two masked-softmax numerators per thread (cols j0+pj, j0+32+pj).
        float e0 = my_es + edb[j0 + pj];
        e0 = e0 > 0.f ? e0 : ALPHA_LRELU * e0;
        float p0 = (arow[j0] > 0) ? __expf(e0 - my_m) : 0.f;
        float e1 = my_es + edb[j0 + 32 + pj];
        e1 = e1 > 0.f ? e1 : ALPHA_LRELU * e1;
        float p1 = (arow[j0 + 32] > 0) ? __expf(e1 - my_m) : 0.f;
        unsigned short* prow = &pTile[set][pr * 80];
        prow[pj]      = f2bf(p0);
        prow[32 + pj] = f2bf(p1);
        if (j0 + 64 < N_) {
            __builtin_prefetch(bcol + j0 + 64 + kbase, 0, 3);
            __builtin_prefetch(arow + j0 + 64, 0, 3);
        }
        __syncthreads();   // single barrier; ping-pong covers the WAR side

        ABPack A0, A1, B0, B1;
        const char* ab = (const char*)(&pTile[set][0]) + lm * 160 + kbase * 2;
        A0.u4[0] = *((const uint4*)ab);          // chunk0: K kbase..+7
        A0.u4[1] = *((const uint4*)(ab + 32));   // chunk0: K kbase+16..+23
        A1.u4[0] = *((const uint4*)(ab + 64));   // chunk1
        A1.u4[1] = *((const uint4*)(ab + 96));
        const unsigned short* bp = bcol + j0 + kbase;
        B0.u4[0] = ((const uint4*)bp)[0];
        B0.u4[1] = ((const uint4*)(bp + 16))[0];
        B1.u4[0] = ((const uint4*)(bp + 32))[0];
        B1.u4[1] = ((const uint4*)(bp + 48))[0];
        acc = __builtin_amdgcn_wmma_f32_16x16x32_bf16(
                  false, A0.v, false, B0.v, (short)0, acc, false, false);
        acc = __builtin_amdgcn_wmma_f32_16x16x32_bf16(
                  false, A1.v, false, B1.v, (short)0, acc, false, false);
    }

#pragma unroll
    for (int r = 0; r < 8; ++r) {
        int   mi = r + 8 * hi;
        float lv = sl[mi];
        float v  = (lv > 0.f) ? acc[r] / lv : 0.f;   // normalize softmax
        v = v > 0.f ? v : (__expf(v) - 1.f);         // ELU (alpha=1)
        out[((size_t)(b * N_ + i0 + mi)) * D_ + ncol] = v;
    }
}

// ---------------------------------------------------------------------------
extern "C" void kernel_launch(void* const* d_in, const int* in_sizes, int n_in,
                              void* d_out, int out_size, void* d_ws, size_t ws_size,
                              hipStream_t stream) {
    const float* h   = (const float*)d_in[0];   // [B,N,Din]
    const int*   adj = (const int*)  d_in[1];   // [N,N]
    const float* W   = (const float*)d_in[2];   // [Din,D]
    const float* a   = (const float*)d_in[3];   // [2D,1]
    float* out = (float*)d_out;

    char* ws = (char*)d_ws;
    size_t off = 0;
    unsigned short* WhbfT = (unsigned short*)(ws + off); off += (size_t)B_ * D_ * N_ * 2; // 8 MB
    unsigned short* hbf   = (unsigned short*)(ws + off); off += (size_t)B_ * N_ * D_ * 2; // 8 MB
    float* e_src = (float*)(ws + off); off += (size_t)B_ * N_ * 4;
    float* e_dst = (float*)(ws + off); off += (size_t)B_ * N_ * 4;
    float* mrow  = (float*)(ws + off); off += (size_t)B_ * N_ * 4;
    float* lrow  = (float*)(ws + off); off += (size_t)B_ * N_ * 4;
    unsigned short* WbfT = (unsigned short*)(ws + off); off += (size_t)D_ * D_ * 2;

    convert_h_kernel<<<(B_ * N_ * D_ / 8) / 256, 256, 0, stream>>>(h, hbf);
    convert_wt_kernel<<<(D_ * D_) / 256, 256, 0, stream>>>(W, WbfT);
    wh_gemm_kernel<<<B_ * (N_ / 16), 512, 0, stream>>>(hbf, WbfT, WhbfT);
    edge_dot_kernel<<<(B_ * N_) / 256, 256, 0, stream>>>(WhbfT, a, e_src, e_dst);
    rowstat_kernel<<<(B_ * N_ * 32) / 256, 256, 0, stream>>>(adj, e_src, e_dst, mrow, lrow);
    gat_aggregate_kernel<<<B_ * (N_ / 16), 512, 0, stream>>>(
        WhbfT, adj, e_src, e_dst, mrow, lrow, out);
}
